// type2_moe_22067541967820
// MI455X (gfx1250) — matrix operand order repaired and checked by reading, not compiled, not run on hardware
//
#include <hip/hip_runtime.h>

typedef __attribute__((ext_vector_type(16))) __bf16 v16bf;
typedef __attribute__((ext_vector_type(8)))  __bf16 v8bf;
typedef __attribute__((ext_vector_type(8)))  float  v8f;

// Problem constants (from reference setup_inputs)
#define HDIM 1024
#define NEXP 3
#define NBR  3
#define TTOK 8192              // B*S = 8*1024 tokens per branch
#define CAP  2731              // ceil(8192/3 * 1.0)
#define LDA  40                // bf16 row stride (80B; keeps 16B-aligned chunks, non-pow2 banks)
#define LDB  40

// ---------------------------------------------------------------------------
// 0) zero the output (dropped tokens must produce zeros)
// ---------------------------------------------------------------------------
__global__ void moe455_zero(float4* __restrict__ out, int n4) {
  int i = blockIdx.x * 256 + threadIdx.x;
  if (i < n4) out[i] = make_float4(0.f, 0.f, 0.f, 0.f);
}

// ---------------------------------------------------------------------------
// 1) gating: logits -> softmax -> top-1 (first max), one wave32 per token
// ---------------------------------------------------------------------------
__global__ __launch_bounds__(256) void moe455_gate(
    const float* __restrict__ x, const float* __restrict__ wg,
    int* __restrict__ idx, float* __restrict__ gate) {
  __shared__ float swg[HDIM * NEXP];  // 12 KB
  int br = blockIdx.y;
  const float* wgb = wg + (size_t)br * HDIM * NEXP;
  for (int i = threadIdx.x; i < HDIM * NEXP; i += 256) swg[i] = wgb[i];
  __syncthreads();

  int wave = threadIdx.x >> 5, lane = threadIdx.x & 31;
  int t = blockIdx.x * 8 + wave;
  size_t base = ((size_t)(t >> 10) * 3 + br) * 1048576u + (size_t)(t & 1023) * HDIM;

  float l0 = 0.f, l1 = 0.f, l2 = 0.f;
  for (int h = lane; h < HDIM; h += 32) {
    float xv = x[base + h];
    l0 = fmaf(xv, swg[h * 3 + 0], l0);
    l1 = fmaf(xv, swg[h * 3 + 1], l1);
    l2 = fmaf(xv, swg[h * 3 + 2], l2);
  }
  for (int off = 16; off > 0; off >>= 1) {
    l0 += __shfl_down(l0, off, 32);
    l1 += __shfl_down(l1, off, 32);
    l2 += __shfl_down(l2, off, 32);
  }
  if (lane == 0) {
    float m = fmaxf(l0, fmaxf(l1, l2));
    float e0 = expf(l0 - m), e1 = expf(l1 - m), e2 = expf(l2 - m);
    float s = e0 + e1 + e2;
    int best = 0; float bv = l0;
    if (l1 > bv) { bv = l1; best = 1; }
    if (l2 > bv) { bv = l2; best = 2; }
    float g = (best == 0 ? e0 : (best == 1 ? e1 : e2)) / s;
    idx[br * TTOK + t] = best;
    gate[br * TTOK + t] = g;
  }
}

// ---------------------------------------------------------------------------
// 2) capacity scan: deterministic token-order cumsum per expert
// ---------------------------------------------------------------------------
__global__ __launch_bounds__(1024) void moe455_scan(
    const int* __restrict__ idx, float* __restrict__ gate,
    int* __restrict__ token_ids) {
  __shared__ int s[NEXP][1024];  // 12 KB
  int br = blockIdx.x, tid = threadIdx.x;
  int* tids = token_ids + br * (NEXP * CAP);
  for (int i = tid; i < NEXP * CAP; i += 1024) tids[i] = -1;

  const int* idxb = idx + br * TTOK;
  int t0 = tid * 8;
  int le[8];
  int c0 = 0, c1 = 0, c2 = 0;
  #pragma unroll
  for (int j = 0; j < 8; j++) {
    int e = idxb[t0 + j];
    le[j] = e;
    if (e == 0) c0++; else if (e == 1) c1++; else c2++;
  }
  s[0][tid] = c0; s[1][tid] = c1; s[2][tid] = c2;
  __syncthreads();
  for (int off = 1; off < 1024; off <<= 1) {
    int v0 = (tid >= off) ? s[0][tid - off] : 0;
    int v1 = (tid >= off) ? s[1][tid - off] : 0;
    int v2 = (tid >= off) ? s[2][tid - off] : 0;
    __syncthreads();
    s[0][tid] += v0; s[1][tid] += v1; s[2][tid] += v2;
    __syncthreads();
  }
  int base[NEXP];
  base[0] = s[0][tid] - c0; base[1] = s[1][tid] - c1; base[2] = s[2][tid] - c2;
  #pragma unroll
  for (int j = 0; j < 8; j++) {
    int e = le[j];
    int loc = base[e]++;
    int t = t0 + j;
    if (loc < CAP) tids[e * CAP + loc] = t;
    else gate[br * TTOK + t] = 0.f;  // dropped token
  }
}

// convert 8 consecutive f32 -> bf16 hi + bf16 lo residual
__device__ __forceinline__ void cvt8_hilo(const float4 u0, const float4 u1,
                                          v8bf& hi, v8bf& lo) {
  float v0 = u0.x, v1 = u0.y, v2 = u0.z, v3 = u0.w;
  float v4 = u1.x, v5 = u1.y, v6 = u1.z, v7 = u1.w;
  __bf16 h;
  h = (__bf16)v0; hi[0] = h; lo[0] = (__bf16)(v0 - (float)h);
  h = (__bf16)v1; hi[1] = h; lo[1] = (__bf16)(v1 - (float)h);
  h = (__bf16)v2; hi[2] = h; lo[2] = (__bf16)(v2 - (float)h);
  h = (__bf16)v3; hi[3] = h; lo[3] = (__bf16)(v3 - (float)h);
  h = (__bf16)v4; hi[4] = h; lo[4] = (__bf16)(v4 - (float)h);
  h = (__bf16)v5; hi[5] = h; lo[5] = (__bf16)(v5 - (float)h);
  h = (__bf16)v6; hi[6] = h; lo[6] = (__bf16)(v6 - (float)h);
  h = (__bf16)v7; hi[7] = h; lo[7] = (__bf16)(v7 - (float)h);
}

#define SHUF16(a, b) __builtin_shufflevector(a, b, 0,1,2,3,4,5,6,7,8,9,10,11,12,13,14,15)
#define WMMA_BF16(A, B, C) \
  __builtin_amdgcn_wmma_f32_16x16x32_bf16(false, A, false, B, (short)0, C, false, false)

// ---------------------------------------------------------------------------
// 3) fused dispatch-gather + expert GEMM (bf16 hi/lo split, f32 accumulate)
//    + bias + gate-scale + combine-scatter.
//    Block = 8 waves; 128(slots) x 64(features) tile; waves 4x2; wave tile
//    32x32 = 4 accumulators -> 12 WMMAs per 16 ds_load_b128 per k-step.
// ---------------------------------------------------------------------------
__global__ __launch_bounds__(256) void moe455_gemm(
    const float* __restrict__ x, const float* __restrict__ We,
    const float* __restrict__ be, const int* __restrict__ token_ids,
    const float* __restrict__ gate, float* __restrict__ out) {
  __shared__ alignas(16) __bf16 Ah[128 * LDA], Al[128 * LDA];  // [row][k]
  __shared__ alignas(16) __bf16 Bh[64 * LDB],  Bl[64 * LDB];   // [n][k] (W natural order)
  __shared__ int      rtok[128];
  __shared__ float    rgate[128];
  __shared__ unsigned rbase[128];  // float-element offset into features (<2^32)

  int bez = blockIdx.z;               // 0..8: branch*3 + expert
  int brch = bez / NEXP, e = bez % NEXP;
  int c0 = blockIdx.y * 128;          // capacity-slot tile
  int n0 = blockIdx.x * 64;           // output-feature tile
  int tid = threadIdx.x;

  const int* tids = token_ids + (brch * NEXP + e) * CAP;
  if (tid < 128) {
    int c = c0 + tid;
    int tok = (c < CAP) ? tids[c] : -1;
    rtok[tid] = tok;
    rgate[tid] = (tok >= 0) ? gate[brch * TTOK + tok] : 0.f;
    unsigned t = (tok >= 0) ? (unsigned)tok : 0u;   // invalid rows alias token 0 (never stored)
    rbase[tid] = ((t >> 10) * 3u + (unsigned)brch) * 1048576u + (t & 1023u) * 1024u;
  }
  __syncthreads();

  const float* wb = We + (size_t)(brch * NEXP + e) * 1048576u;  // [o][h]
  v8f acc00 = {}, acc01 = {}, acc10 = {}, acc11 = {};
  int lane = tid & 31, wave = tid >> 5;
  int wm = wave & 3, wn = wave >> 2;
  int l15 = lane & 15, hs = lane >> 4;

  // A loader: thread owns 16 consecutive floats: row ra = tid>>1, half qa = tid&1
  int ra = tid >> 1, qa = tid & 1;
  // B loader: thread owns 8 consecutive floats: row rb = tid>>2, quarter qb = tid&3
  int rb = tid >> 2, qb = tid & 3;
  const float* srcB = wb + (size_t)(n0 + rb) * HDIM + qb * 8;

  for (int k0 = 0; k0 < HDIM; k0 += 32) {
    __syncthreads();
    {
      // A tile: 128x32 gathered token rows -> bf16 hi/lo
      const float* srcA = x + (size_t)rbase[ra] + (unsigned)(k0 + qa * 16);
      float4 a0 = ((const float4*)srcA)[0];
      float4 a1 = ((const float4*)srcA)[1];
      float4 a2 = ((const float4*)srcA)[2];
      float4 a3 = ((const float4*)srcA)[3];
      // B tile: 64x32 weight rows in natural [o][h] order
      float4 b0 = ((const float4*)(srcB + k0))[0];
      float4 b1 = ((const float4*)(srcB + k0))[1];
      if (k0 + 32 < HDIM) {                    // prefetch next k-slab
        __builtin_prefetch(srcA + 32, 0, 1);
        __builtin_prefetch(srcB + k0 + 32, 0, 1);
      }
      v8bf hi, lo;
      cvt8_hilo(a0, a1, hi, lo);
      *(v8bf*)(Ah + ra * LDA + qa * 16) = hi;
      *(v8bf*)(Al + ra * LDA + qa * 16) = lo;
      cvt8_hilo(a2, a3, hi, lo);
      *(v8bf*)(Ah + ra * LDA + qa * 16 + 8) = hi;
      *(v8bf*)(Al + ra * LDA + qa * 16 + 8) = lo;
      cvt8_hilo(b0, b1, hi, lo);
      *(v8bf*)(Bh + rb * LDB + qb * 8) = hi;
      *(v8bf*)(Bl + rb * LDB + qb * 8) = lo;
    }
    __syncthreads();

    // A fragments (two 16-row subtiles): lane half hs -> K {hs*8..}, {16+hs*8..}
    int ar0 = (wm * 32 + l15) * LDA;
    int ar1 = ar0 + 16 * LDA;
    v16bf afh0 = SHUF16(*(const v8bf*)(Ah + ar0 + hs * 8),
                        *(const v8bf*)(Ah + ar0 + 16 + hs * 8));
    v16bf afl0 = SHUF16(*(const v8bf*)(Al + ar0 + hs * 8),
                        *(const v8bf*)(Al + ar0 + 16 + hs * 8));
    v16bf afh1 = SHUF16(*(const v8bf*)(Ah + ar1 + hs * 8),
                        *(const v8bf*)(Ah + ar1 + 16 + hs * 8));
    v16bf afl1 = SHUF16(*(const v8bf*)(Al + ar1 + hs * 8),
                        *(const v8bf*)(Al + ar1 + 16 + hs * 8));
    // B fragments (two 16-col subtiles): N-major -> lane half hs -> K {hs*8..}, {16+hs*8..}
    int nr0 = (wn * 32 + l15) * LDB + hs * 8;
    int nr1 = nr0 + 16 * LDB;
    v16bf bh0 = SHUF16(*(const v8bf*)(Bh + nr0), *(const v8bf*)(Bh + nr0 + 16));
    v16bf bl0 = SHUF16(*(const v8bf*)(Bl + nr0), *(const v8bf*)(Bl + nr0 + 16));
    v16bf bh1 = SHUF16(*(const v8bf*)(Bh + nr1), *(const v8bf*)(Bh + nr1 + 16));
    v16bf bl1 = SHUF16(*(const v8bf*)(Bl + nr1), *(const v8bf*)(Bl + nr1 + 16));

    // split-precision accumulate: ah*bh + ah*bl + al*bh (al*bl ~ 2^-16, dropped)
    acc00 = WMMA_BF16(afh0, bh0, acc00);
    acc00 = WMMA_BF16(afh0, bl0, acc00);
    acc00 = WMMA_BF16(afl0, bh0, acc00);
    acc01 = WMMA_BF16(afh0, bh1, acc01);
    acc01 = WMMA_BF16(afh0, bl1, acc01);
    acc01 = WMMA_BF16(afl0, bh1, acc01);
    acc10 = WMMA_BF16(afh1, bh0, acc10);
    acc10 = WMMA_BF16(afh1, bl0, acc10);
    acc10 = WMMA_BF16(afl1, bh0, acc10);
    acc11 = WMMA_BF16(afh1, bh1, acc11);
    acc11 = WMMA_BF16(afh1, bl1, acc11);
    acc11 = WMMA_BF16(afl1, bh1, acc11);
  }

  // epilogue: out[token] = gate * (acc + bias); C/D layout: VGPR i -> M = i + 8*hs
  const float* bias = be + (size_t)(brch * NEXP + e) * HDIM;
  int ncol0 = n0 + wn * 32 + l15;
  int ncol1 = ncol0 + 16;
  float b0 = bias[ncol0], b1 = bias[ncol1];
  #pragma unroll
  for (int i = 0; i < 8; i++) {
    int rl0 = wm * 32 + i + hs * 8;         // rows of acc0x
    int rl1 = rl0 + 16;                     // rows of acc1x
    int tok = rtok[rl0];
    if (tok >= 0) {
      float g = rgate[rl0];
      unsigned ob = ((unsigned)(tok >> 10) * 3u + (unsigned)brch) * 1048576u +
                    (unsigned)(tok & 1023) * 1024u;
      out[ob + ncol0] = g * (acc00[i] + b0);
      out[ob + ncol1] = g * (acc01[i] + b1);
    }
    tok = rtok[rl1];
    if (tok >= 0) {
      float g = rgate[rl1];
      unsigned ob = ((unsigned)(tok >> 10) * 3u + (unsigned)brch) * 1048576u +
                    (unsigned)(tok & 1023) * 1024u;
      out[ob + ncol0] = g * (acc10[i] + b0);
      out[ob + ncol1] = g * (acc11[i] + b1);
    }
  }
}

// ---------------------------------------------------------------------------
extern "C" void kernel_launch(void* const* d_in, const int* in_sizes, int n_in,
                              void* d_out, int out_size, void* d_ws, size_t ws_size,
                              hipStream_t stream) {
  const float* features = (const float*)d_in[0];  // [8,3,1024,1024]
  const float* gate_w   = (const float*)d_in[1];  // [3,1024,3]
  const float* expert_w = (const float*)d_in[2];  // [3,3,1024,1024]
  const float* expert_b = (const float*)d_in[3];  // [3,3,1024]
  float* out = (float*)d_out;

  // workspace: idx[3*8192] i32 | gate[3*8192] f32 | token_ids[3*3*2731] i32
  int*   ws_idx  = (int*)d_ws;
  float* ws_gate = (float*)((char*)d_ws + (size_t)NBR * TTOK * 4);
  int*   ws_tids = (int*)((char*)d_ws + (size_t)NBR * TTOK * 8);

  moe455_zero<<<24576, 256, 0, stream>>>((float4*)out, 6291456);
  moe455_gate<<<dim3(1024, 3), 256, 0, stream>>>(features, gate_w, ws_idx, ws_gate);
  moe455_scan<<<3, 1024, 0, stream>>>(ws_idx, ws_gate, ws_tids);
  // grid: 16 n-tiles x ceil(2731/128)=22 c-tiles x (3 branches * 3 experts)
  moe455_gemm<<<dim3(16, 22, 9), 256, 0, stream>>>(
      features, expert_w, expert_b, ws_tids, ws_gate, out);
}
